// RNN_64072322122514
// MI455X (gfx1250) — matrix operand order repaired and checked by reading, not compiled, not run on hardware
//
#include <hip/hip_runtime.h>

// CDNA5 / gfx1250 RNN scan: h_{t+1} = tanh(x_t @ W_ih^T + h_t @ W_hh^T + b)
// Each wave owns TWO independent 16-row batch tiles and interleaves their
// WMMA/tanh/LDS-transpose chains to hide the serial per-step latency
// (the recurrence is latency-bound: only B/16 = 256 independent chains).
// The bias is folded into the input-projection WMMA via an augmented K=16
// column (A_x[:,16] = 1, B_x[16,:] = b_ih+b_hh), so the accumulator chain
// starts from a loop-invariant zero C block (no per-step re-init).

typedef _Float16 v16h __attribute__((ext_vector_type(16)));
typedef float    v8f  __attribute__((ext_vector_type(8)));
typedef float    v4f  __attribute__((ext_vector_type(4)));

#define BB 4096
#define TT 2048
#define FF 8
#define HH 32
#define OO 8

__device__ __forceinline__ float fast_tanh(float v) {
#if __has_builtin(__builtin_amdgcn_tanhf)
    return __builtin_amdgcn_tanhf(v);     // gfx1250 v_tanh_f32 (trans op)
#else
    float e = __expf(2.0f * v);
    return 1.0f - 2.0f / (e + 1.0f);
#endif
}

__global__ __launch_bounds__(32) void rnn_scan_kernel(
    const float* __restrict__ x,     // [B, T, F]
    const float* __restrict__ W_ih,  // [H, F]
    const float* __restrict__ W_hh,  // [H, H]
    const float* __restrict__ b_ih,  // [H]
    const float* __restrict__ b_hh,  // [H]
    const float* __restrict__ W_out, // [O, H]
    const float* __restrict__ b_out, // [O]
    float* __restrict__ y)           // [1, B, O]
{
    __shared__ float hbuf[2][16 * 32];   // one h tile per chain, [m][n] f32

    const int lane = threadIdx.x;    // 0..31
    const int lo   = lane & 15;
    const int hi   = lane >> 4;      // 0 or 1
    const int b0   = blockIdx.x * 32;   // 32 batch rows per wave (2 tiles)

    // --- constant operands (held in VGPRs across the whole scan) -----------
    // B operand k order used here (contiguous, same as A): element 2v+h of v16h lives in
    // VGPR v half h;  K = 2v + 16*hi + h,  N = lo (+16 for second N-tile).
    v16h bh0, bh1, bx0, bx1;
#pragma unroll
    for (int v = 0; v < 8; ++v) {
        const int k = 2 * v + 16 * hi;
        bh0[2*v]   = (_Float16)W_hh[lo * HH + k];
        bh0[2*v+1] = (_Float16)W_hh[lo * HH + k + 1];
        bh1[2*v]   = (_Float16)W_hh[(lo + 16) * HH + k];
        bh1[2*v+1] = (_Float16)W_hh[(lo + 16) * HH + k + 1];
        if (hi == 0 && v < 4) {      // W_ih rows K=0..7 (lo-lanes only)
            bx0[2*v]   = (_Float16)W_ih[lo * FF + 2 * v];
            bx0[2*v+1] = (_Float16)W_ih[lo * FF + 2 * v + 1];
            bx1[2*v]   = (_Float16)W_ih[(lo + 16) * FF + 2 * v];
            bx1[2*v+1] = (_Float16)W_ih[(lo + 16) * FF + 2 * v + 1];
        } else {
            bx0[2*v] = (_Float16)0.f; bx0[2*v+1] = (_Float16)0.f;
            bx1[2*v] = (_Float16)0.f; bx1[2*v+1] = (_Float16)0.f;
        }
    }
    // Augmented bias row at K=16: element 0 on hi-lanes (K = 0 + 16*1).
    if (hi == 1) {
        bx0[0] = (_Float16)(b_ih[lo]      + b_hh[lo]);
        bx1[0] = (_Float16)(b_ih[lo + 16] + b_hh[lo + 16]);
    }

    // Loop-invariant zero C block: first WMMA of each chain reads it as src2.
    v8f czero;
#pragma unroll
    for (int r = 0; r < 8; ++r) czero[r] = 0.0f;

    // Per-chain x row streams (hi-lanes mirror lo-lanes; their A_x half is
    // the constant one-hot 1.0 at K=16 instead of x data).
    const float* px0 = x + (size_t)(b0 + lo)      * TT * FF;
    const float* px1 = x + (size_t)(b0 + 16 + lo) * TT * FF;

    v16h ah0, ah1;                       // h operands (A layout), start at 0
#pragma unroll
    for (int i = 0; i < 16; ++i) { ah0[i] = (_Float16)0.f; ah1[i] = (_Float16)0.f; }

    const _Float16 one_hi  = (hi == 0) ? (_Float16)0.f : (_Float16)1.0f; // K=16 slot
    const bool     is_lo   = (hi == 0);

    // --- the scan: two interleaved independent chains ----------------------
    for (int t = 0; t < TT; ++t) {
        const v4f xa0 = __builtin_nontemporal_load((const v4f*)px0 + 0);
        const v4f xb0 = __builtin_nontemporal_load((const v4f*)px0 + 1);
        const v4f xa1 = __builtin_nontemporal_load((const v4f*)px1 + 0);
        const v4f xb1 = __builtin_nontemporal_load((const v4f*)px1 + 1);
        px0 += FF; px1 += FF;

        // A_x: lo-lanes elements 0..7 = x (K=0..7); hi-lanes element 0 = 1.0
        // (K=16, matches bias row of B_x); everything else zero.
        v16h ax0, ax1;
        ax0[0] = is_lo ? (_Float16)xa0[0] : one_hi;
        ax0[1] = is_lo ? (_Float16)xa0[1] : (_Float16)0.f;
        ax0[2] = is_lo ? (_Float16)xa0[2] : (_Float16)0.f;
        ax0[3] = is_lo ? (_Float16)xa0[3] : (_Float16)0.f;
        ax0[4] = is_lo ? (_Float16)xb0[0] : (_Float16)0.f;
        ax0[5] = is_lo ? (_Float16)xb0[1] : (_Float16)0.f;
        ax0[6] = is_lo ? (_Float16)xb0[2] : (_Float16)0.f;
        ax0[7] = is_lo ? (_Float16)xb0[3] : (_Float16)0.f;
        ax1[0] = is_lo ? (_Float16)xa1[0] : one_hi;
        ax1[1] = is_lo ? (_Float16)xa1[1] : (_Float16)0.f;
        ax1[2] = is_lo ? (_Float16)xa1[2] : (_Float16)0.f;
        ax1[3] = is_lo ? (_Float16)xa1[3] : (_Float16)0.f;
        ax1[4] = is_lo ? (_Float16)xb1[0] : (_Float16)0.f;
        ax1[5] = is_lo ? (_Float16)xb1[1] : (_Float16)0.f;
        ax1[6] = is_lo ? (_Float16)xb1[2] : (_Float16)0.f;
        ax1[7] = is_lo ? (_Float16)xb1[3] : (_Float16)0.f;
#pragma unroll
        for (int i = 8; i < 16; ++i) { ax0[i] = (_Float16)0.f; ax1[i] = (_Float16)0.f; }

        // Interleaved WMMA chains (fresh dst, zero C is read-only src2):
        //   c = A_x*[W_ih;bias] + 0 ;  c = A_h*W_hh + c
        v8f c00, c01, c10, c11;
        c00 = __builtin_amdgcn_wmma_f32_16x16x32_f16(false, ax0, false, bx0,
                                                     (short)0, czero, false, false);
        c10 = __builtin_amdgcn_wmma_f32_16x16x32_f16(false, ax1, false, bx0,
                                                     (short)0, czero, false, false);
        c01 = __builtin_amdgcn_wmma_f32_16x16x32_f16(false, ax0, false, bx1,
                                                     (short)0, czero, false, false);
        c11 = __builtin_amdgcn_wmma_f32_16x16x32_f16(false, ax1, false, bx1,
                                                     (short)0, czero, false, false);
        c00 = __builtin_amdgcn_wmma_f32_16x16x32_f16(false, ah0, false, bh0,
                                                     (short)0, c00, false, false);
        c10 = __builtin_amdgcn_wmma_f32_16x16x32_f16(false, ah1, false, bh0,
                                                     (short)0, c10, false, false);
        c01 = __builtin_amdgcn_wmma_f32_16x16x32_f16(false, ah0, false, bh1,
                                                     (short)0, c01, false, false);
        c11 = __builtin_amdgcn_wmma_f32_16x16x32_f16(false, ah1, false, bh1,
                                                     (short)0, c11, false, false);

        // tanh (gfx1250 v_tanh_f32), scatter D layout to LDS:
        // m = r + 8*hi, n = lo (c*0) / lo+16 (c*1).
#pragma unroll
        for (int r = 0; r < 8; ++r) {
            hbuf[0][(hi * 8 + r) * 32 + lo]      = fast_tanh(c00[r]);
            hbuf[0][(hi * 8 + r) * 32 + 16 + lo] = fast_tanh(c01[r]);
            hbuf[1][(hi * 8 + r) * 32 + lo]      = fast_tanh(c10[r]);
            hbuf[1][(hi * 8 + r) * 32 + 16 + lo] = fast_tanh(c11[r]);
        }
        __syncthreads();   // single-wave WG: compiler fence + dscnt wait

        // Gather A layout: m = lo across lanes, K pairs across VGPR halves.
        // K(v,hi) = 2v + 16*hi  (even -> aligned float2)
#pragma unroll
        for (int v = 0; v < 8; ++v) {
            const int k0 = 2 * v + 16 * hi;
            const float2 h0 = *(const float2*)&hbuf[0][lo * 32 + k0];
            const float2 h1 = *(const float2*)&hbuf[1][lo * 32 + k0];
            ah0[2*v]   = (_Float16)h0.x;  ah0[2*v+1] = (_Float16)h0.y;
            ah1[2*v]   = (_Float16)h1.x;  ah1[2*v+1] = (_Float16)h1.y;
        }
        __syncthreads();   // WAR guard before next iteration's stores
    }

    // --- tiny output head: y = h_final @ W_out^T + b_out -------------------
    // Lane (lo,hi) computes y[row, hi*4 .. hi*4+3] for each tile's row lo.
    const float* wo = W_out + (size_t)(hi * 4) * HH;
#pragma unroll
    for (int j = 0; j < 2; ++j) {
        float acc0 = b_out[hi * 4 + 0];
        float acc1 = b_out[hi * 4 + 1];
        float acc2 = b_out[hi * 4 + 2];
        float acc3 = b_out[hi * 4 + 3];
#pragma unroll
        for (int n = 0; n < HH; ++n) {
            const float hv = hbuf[j][lo * 32 + n];
            acc0 += hv * wo[0 * HH + n];
            acc1 += hv * wo[1 * HH + n];
            acc2 += hv * wo[2 * HH + n];
            acc3 += hv * wo[3 * HH + n];
        }
        float4 o4 = make_float4(acc0, acc1, acc2, acc3);
        *(float4*)&y[(size_t)(b0 + j * 16 + lo) * OO + hi * 4] = o4;
    }
}

extern "C" void kernel_launch(void* const* d_in, const int* in_sizes, int n_in,
                              void* d_out, int out_size, void* d_ws, size_t ws_size,
                              hipStream_t stream) {
    (void)in_sizes; (void)n_in; (void)out_size; (void)d_ws; (void)ws_size;
    const float* x     = (const float*)d_in[0];
    const float* W_ih  = (const float*)d_in[1];
    const float* W_hh  = (const float*)d_in[2];
    const float* b_ih  = (const float*)d_in[3];
    const float* b_hh  = (const float*)d_in[4];
    const float* W_out = (const float*)d_in[5];
    const float* b_out = (const float*)d_in[6];
    float* y = (float*)d_out;

    dim3 grid(BB / 32);   // 128 waves, each running 2 interleaved tile chains
    dim3 block(32);       // one wave32 per workgroup -> barriers are free
    rnn_scan_kernel<<<grid, block, 0, stream>>>(x, W_ih, W_hh, b_ih, b_hh,
                                                W_out, b_out, y);
}
